// AtomTransformerCS_13718125543679
// MI455X (gfx1250) — compile-verified, matrix operand order checked by reading, not run on hardware
//
#include <hip/hip_runtime.h>
#include <hip/hip_bf16.h>
#include <math.h>

// ---- problem sizes ----
#define BI   4
#define NI   512
#define EI   64
#define HDI  512
#define NHI  8
#define LI   6
#define NBI  4
#define NKI  64
#define DHI  64
#define MI   (BI * NI)   // 2048 tokens
#define FFI  (4 * HDI)   // 2048
#define HHI  (HDI / 2)   // 256

typedef __bf16 bf16;
typedef __attribute__((ext_vector_type(16))) __bf16 v16bf;
typedef __attribute__((ext_vector_type(8)))  float  v8f;

// ---------------------------------------------------------------------------
// WMMA tile loader (CDNA5 16-bit A-matrix 16x32 layout, wave32):
// lane L: row = L&15, half = L>>4; elems 0..7 -> K = half*8 + e,
// elems 8..15 -> K = 16 + half*8 + (e-8).   Two b128 loads per lane.
// ---------------------------------------------------------------------------
__device__ inline v16bf load_tile_contig(const bf16* __restrict__ base, int rs, int lane) {
    const int row  = lane & 15;
    const int half = lane >> 4;
    const bf16* p = base + (long)row * rs + half * 8;
    v16bf r;
#pragma unroll
    for (int e = 0; e < 8; ++e) { r[e] = p[e]; r[e + 8] = p[16 + e]; }
    return r;
}

__device__ inline float gelu_f(float x) {
    return 0.5f * x * (1.0f + erff(x * 0.70710678118654752f));
}

// ---------------------------------------------------------------------------
// Dense GEMM, 32x64 macro-tile per wave (2 A-tiles x 4 B-tiles, 8 acc):
// C_bf16[M,N] = act(A_bf16[M,K] @ Bt_bf16[N,K]^T + bias[N])
// ---------------------------------------------------------------------------
__global__ void gemm_bf16out(const bf16* __restrict__ A, const bf16* __restrict__ Bt,
                             const float* __restrict__ bias, bf16* __restrict__ C,
                             int M, int N, int K, int do_gelu) {
    const int lane = threadIdx.x & 31;
    const int wave = threadIdx.x >> 5;
    const int ntN  = N >> 6;                   // 64-wide column strips
    const int ntiles = (M >> 5) * ntN;         // 32-tall row strips
    const int tile = blockIdx.x * 8 + wave;
    if (tile >= ntiles) return;
    const int tm = (tile / ntN) << 5;
    const int tn = (tile % ntN) << 6;

    v8f acc[2][4] = {};
    for (int k0 = 0; k0 < K; k0 += 32) {
        v16bf a0 = load_tile_contig(A + (long)tm * K + k0, K, lane);
        v16bf a1 = load_tile_contig(A + (long)(tm + 16) * K + k0, K, lane);
#pragma unroll
        for (int ni = 0; ni < 4; ++ni) {
            v16bf bv = load_tile_contig(Bt + (long)(tn + ni * 16) * K + k0, K, lane);
            acc[0][ni] = __builtin_amdgcn_wmma_f32_16x16x32_bf16(false, a0, false, bv, (short)0, acc[0][ni], false, false);
            acc[1][ni] = __builtin_amdgcn_wmma_f32_16x16x32_bf16(false, a1, false, bv, (short)0, acc[1][ni], false, false);
        }
    }
    const int col  = lane & 15;
    const int half = lane >> 4;
#pragma unroll
    for (int ni = 0; ni < 4; ++ni) {
        const float bb = bias[tn + ni * 16 + col];
#pragma unroll
        for (int mi = 0; mi < 2; ++mi) {
#pragma unroll
            for (int r = 0; r < 8; ++r) {
                const int row = tm + mi * 16 + r + 8 * half;
                float v = acc[mi][ni][r] + bb;
                if (do_gelu) v = gelu_f(v);
                C[(long)row * N + tn + ni * 16 + col] = (bf16)v;
            }
        }
    }
}

// ---------------------------------------------------------------------------
// Residual GEMM, 32x64 macro-tile: X_f32[M,N] += (A@Bt^T + bias) * rowmask[m]
// ---------------------------------------------------------------------------
__global__ void gemm_f32resid(const bf16* __restrict__ A, const bf16* __restrict__ Bt,
                              const float* __restrict__ bias, float* __restrict__ X,
                              const unsigned char* __restrict__ rowmask,
                              int M, int N, int K) {
    const int lane = threadIdx.x & 31;
    const int wave = threadIdx.x >> 5;
    const int ntN  = N >> 6;
    const int ntiles = (M >> 5) * ntN;
    const int tile = blockIdx.x * 8 + wave;
    if (tile >= ntiles) return;
    const int tm = (tile / ntN) << 5;
    const int tn = (tile % ntN) << 6;

    v8f acc[2][4] = {};
    for (int k0 = 0; k0 < K; k0 += 32) {
        v16bf a0 = load_tile_contig(A + (long)tm * K + k0, K, lane);
        v16bf a1 = load_tile_contig(A + (long)(tm + 16) * K + k0, K, lane);
#pragma unroll
        for (int ni = 0; ni < 4; ++ni) {
            v16bf bv = load_tile_contig(Bt + (long)(tn + ni * 16) * K + k0, K, lane);
            acc[0][ni] = __builtin_amdgcn_wmma_f32_16x16x32_bf16(false, a0, false, bv, (short)0, acc[0][ni], false, false);
            acc[1][ni] = __builtin_amdgcn_wmma_f32_16x16x32_bf16(false, a1, false, bv, (short)0, acc[1][ni], false, false);
        }
    }
    const int col  = lane & 15;
    const int half = lane >> 4;
#pragma unroll
    for (int ni = 0; ni < 4; ++ni) {
        const float bb = bias[tn + ni * 16 + col];
#pragma unroll
        for (int mi = 0; mi < 2; ++mi) {
#pragma unroll
            for (int r = 0; r < 8; ++r) {
                const int row = tm + mi * 16 + r + 8 * half;
                float mk = 1.0f;
                if (rowmask) mk = rowmask[row] ? 1.0f : 0.0f;
                X[(long)row * N + tn + ni * 16 + col] += (acc[mi][ni][r] + bb) * mk;
            }
        }
    }
}

// ---------------------------------------------------------------------------
// Attention scores, 32x64 macro-tile per wave, batched over z=(b,h):
// scores[z,i,j] = scale * q[b,i,h,:] . k[b,j,h,:]
// ---------------------------------------------------------------------------
__global__ void attn_scores(const bf16* __restrict__ q, const bf16* __restrict__ k,
                            float* __restrict__ scores, float scale) {
    const int z = blockIdx.y;               // b*NH + h
    const int b = z >> 3, h = z & 7;
    const int lane = threadIdx.x & 31;
    const int wave = threadIdx.x >> 5;
    const int tile = blockIdx.x * 8 + wave; // (512/32)*(512/64) = 128 tiles
    if (tile >= (NI / 32) * (NI / 64)) return;
    const int ti = (tile >> 3) << 5;        // 16 row-strips
    const int tj = (tile & 7) << 6;         // 8 col-strips

    const bf16* qb = q + (long)b * NI * HDI + h * DHI;
    const bf16* kb = k + (long)b * NI * HDI + h * DHI;
    v8f acc[2][4] = {};
    for (int k0 = 0; k0 < DHI; k0 += 32) {
        v16bf a0 = load_tile_contig(qb + (long)ti * HDI + k0, HDI, lane);
        v16bf a1 = load_tile_contig(qb + (long)(ti + 16) * HDI + k0, HDI, lane);
#pragma unroll
        for (int ni = 0; ni < 4; ++ni) {
            v16bf bv = load_tile_contig(kb + (long)(tj + ni * 16) * HDI + k0, HDI, lane);
            acc[0][ni] = __builtin_amdgcn_wmma_f32_16x16x32_bf16(false, a0, false, bv, (short)0, acc[0][ni], false, false);
            acc[1][ni] = __builtin_amdgcn_wmma_f32_16x16x32_bf16(false, a1, false, bv, (short)0, acc[1][ni], false, false);
        }
    }
    const int col  = lane & 15;
    const int half = lane >> 4;
    float* out = scores + (long)z * NI * NI;
#pragma unroll
    for (int ni = 0; ni < 4; ++ni)
#pragma unroll
        for (int mi = 0; mi < 2; ++mi)
#pragma unroll
            for (int r = 0; r < 8; ++r)
                out[(long)(ti + mi * 16 + r + 8 * half) * NI + tj + ni * 16 + col] = acc[mi][ni][r] * scale;
}

// ---------------------------------------------------------------------------
// V transpose: vt[z=(b,h)][d][j] = v[b,j,h,d]  (contiguous Bt for attn_out)
// ---------------------------------------------------------------------------
__global__ void vt_k(const bf16* __restrict__ v, bf16* __restrict__ vt) {
    const int t = blockIdx.x * 256 + threadIdx.x;   // B*NH*DH*N = 1048576
    const int j = t & 511;
    const int d = (t >> 9) & 63;
    const int z = t >> 15;
    const int b = z >> 3, h = z & 7;
    vt[t] = v[((long)(b * NI + j)) * HDI + h * DHI + d];
}

// ---------------------------------------------------------------------------
// Attention output: o[b,i,h,d] = sum_j attn[z,i,j] * vt[z,d,j]  (bf16 out)
// 32x64 macro-tile (full head dim in one strip), batched over z
// ---------------------------------------------------------------------------
__global__ void attn_out(const bf16* __restrict__ attn, const bf16* __restrict__ vt,
                         bf16* __restrict__ o) {
    const int z = blockIdx.y;
    const int b = z >> 3, h = z & 7;
    const int lane = threadIdx.x & 31;
    const int wave = threadIdx.x >> 5;
    const int tile = blockIdx.x * 8 + wave;  // 512/32 = 16 row-strips
    if (tile >= NI / 32) return;
    const int ti = tile << 5;

    const bf16* ab = attn + (long)z * NI * NI;
    const bf16* vb = vt + (long)z * DHI * NI;
    v8f acc[2][4] = {};
    for (int j0 = 0; j0 < NI; j0 += 32) {
        v16bf a0 = load_tile_contig(ab + (long)ti * NI + j0, NI, lane);
        v16bf a1 = load_tile_contig(ab + (long)(ti + 16) * NI + j0, NI, lane);
#pragma unroll
        for (int ni = 0; ni < 4; ++ni) {
            v16bf bv = load_tile_contig(vb + (long)(ni * 16) * NI + j0, NI, lane);
            acc[0][ni] = __builtin_amdgcn_wmma_f32_16x16x32_bf16(false, a0, false, bv, (short)0, acc[0][ni], false, false);
            acc[1][ni] = __builtin_amdgcn_wmma_f32_16x16x32_bf16(false, a1, false, bv, (short)0, acc[1][ni], false, false);
        }
    }
    const int col  = lane & 15;
    const int half = lane >> 4;
#pragma unroll
    for (int ni = 0; ni < 4; ++ni)
#pragma unroll
        for (int mi = 0; mi < 2; ++mi)
#pragma unroll
            for (int r = 0; r < 8; ++r) {
                const int row = ti + mi * 16 + r + 8 * half;
                o[((long)(b * NI + row)) * HDI + h * DHI + ni * 16 + col] = (bf16)acc[mi][ni][r];
            }
}

// ---------------------------------------------------------------------------
// Gaussian RBF distance bias: scores[b,h,i,j] += bd[h] + sum_k rbf_k(d) * Wd[k,h]
// ---------------------------------------------------------------------------
__global__ void bias_add(const float* __restrict__ dist, const float* __restrict__ centers,
                         const float* __restrict__ widths, const float* __restrict__ Wd,
                         const float* __restrict__ bd, float* __restrict__ scores) {
    __shared__ float sWd[NKI * NHI];
    __shared__ float sC[NKI];
    __shared__ float sW[NKI];
    const int t = threadIdx.x;
    sWd[t] = Wd[t]; sWd[t + 256] = Wd[t + 256];
    if (t < NKI) { sC[t] = centers[t]; sW[t] = widths[t]; }
    __syncthreads();

    const long idx = (long)blockIdx.x * 256 + t;   // over B*N*N = 1048576
    const int b = (int)(idx >> 18);
    const int rem = (int)(idx & 262143);
    const int i = rem >> 9;
    const int j = rem & 511;

    float d = dist[idx];
    d = fminf(fmaxf(d, 0.0f), 20.0f);

    float acc[NHI];
#pragma unroll
    for (int h = 0; h < NHI; ++h) acc[h] = bd[h];

    for (int k = 0; k < NKI; ++k) {
        float w = fminf(fmaxf(fabsf(sW[k]), 0.1f), 5.0f);
        float dc = d - sC[k];
        float e = __expf(-dc * dc / (2.0f * w * w));
        e = fminf(fmaxf(e, 1e-8f), 1.0f);
#pragma unroll
        for (int h = 0; h < NHI; ++h) acc[h] += e * sWd[k * NHI + h];
    }
#pragma unroll
    for (int h = 0; h < NHI; ++h)
        scores[(((long)(b * NHI + h) * NI) + i) * NI + j] += acc[h];
}

// ---------------------------------------------------------------------------
// Row softmax with key/query masking; writes bf16 attn.  blockDim = 512 = N
// ---------------------------------------------------------------------------
__global__ void softmax_k(const float* __restrict__ scores, const unsigned char* __restrict__ mask,
                          bf16* __restrict__ attn) {
    const int row = blockIdx.x;         // z*N + i, z = b*NH+h
    const int j = threadIdx.x;
    const int z = row >> 9;
    const int i = row & 511;
    const int b = z >> 3;

    const float s = scores[(long)row * NI + j];
    const bool mj = mask[b * NI + j] != 0;
    const float val = mj ? s : -1e9f;

    __shared__ float red[NI];
    red[j] = val;
    __syncthreads();
    for (int st = 256; st > 0; st >>= 1) {
        if (j < st) red[j] = fmaxf(red[j], red[j + st]);
        __syncthreads();
    }
    const float mx = red[0];
    __syncthreads();
    const float e = __expf(val - mx);
    red[j] = e;
    __syncthreads();
    for (int st = 256; st > 0; st >>= 1) {
        if (j < st) red[j] += red[j + st];
        __syncthreads();
    }
    const float sum = red[0];
    const float mi = mask[b * NI + i] ? 1.0f : 0.0f;
    attn[(long)row * NI + j] = (bf16)(e / sum * mi);
}

// ---------------------------------------------------------------------------
// LayerNorm: out_bf16 = LN(x) * g + b   (one block of 512 threads per token)
// ---------------------------------------------------------------------------
__global__ void ln_k(const float* __restrict__ x, const float* __restrict__ g,
                     const float* __restrict__ be, bf16* __restrict__ out) {
    const int m = blockIdx.x;
    const int t = threadIdx.x;
    const float v = x[(long)m * HDI + t];
    __shared__ float red[HDI];
    red[t] = v;
    __syncthreads();
    for (int st = 256; st > 0; st >>= 1) {
        if (t < st) red[t] += red[t + st];
        __syncthreads();
    }
    const float mean = red[0] * (1.0f / HDI);
    __syncthreads();
    const float d = v - mean;
    red[t] = d * d;
    __syncthreads();
    for (int st = 256; st > 0; st >>= 1) {
        if (t < st) red[t] += red[t + st];
        __syncthreads();
    }
    const float var = red[0] * (1.0f / HDI);
    out[(long)m * HDI + t] = (bf16)(d * rsqrtf(var + 1e-5f) * g[t] + be[t]);
}

// ---------------------------------------------------------------------------
// Embedding + concat + input projection (384->512) + LN -> x_f32
// ---------------------------------------------------------------------------
__global__ void embed_k(const int* atype, const int* aname, const int* rtype,
                        const int* sstype, const int* relpos,
                        const float* coords, const float* phi, const float* psi,
                        const float* cs,
                        const float* Eat, const float* Ean, const float* Er,
                        const float* Ess, const float* Ep,
                        const float* Wc, const float* bc,
                        const float* Win, const float* bin,
                        const float* gin, const float* bein,
                        float* __restrict__ x) {
    const int m = blockIdx.x;      // token
    const int t = threadIdx.x;     // 512
    __shared__ float cc[6 * EI];   // 384
    if (t < 6 * EI) {
        const int seg = t >> 6, j = t & 63;
        float v;
        if (seg == 0)      v = Eat[atype[m] * EI + j];
        else if (seg == 1) v = Ean[aname[m] * EI + j];
        else if (seg == 2) v = Er[rtype[m] * EI + j];
        else if (seg == 3) v = Ess[sstype[m] * EI + j];
        else if (seg == 4) v = Ep[(relpos[m] + 10) * EI + j];
        else {
            v = bc[j]
              + coords[(long)m * 3 + 0] * Wc[0 * EI + j]
              + coords[(long)m * 3 + 1] * Wc[1 * EI + j]
              + coords[(long)m * 3 + 2] * Wc[2 * EI + j]
              + phi[m] * Wc[3 * EI + j]
              + psi[m] * Wc[4 * EI + j]
              + cs[m]  * Wc[5 * EI + j];
        }
        cc[t] = v;
    }
    __syncthreads();
    float y = bin[t];
    for (int k = 0; k < 6 * EI; ++k) y += cc[k] * Win[(long)k * HDI + t];

    __shared__ float red[HDI];
    red[t] = y;
    __syncthreads();
    for (int st = 256; st > 0; st >>= 1) {
        if (t < st) red[t] += red[t + st];
        __syncthreads();
    }
    const float mean = red[0] * (1.0f / HDI);
    __syncthreads();
    const float d = y - mean;
    red[t] = d * d;
    __syncthreads();
    for (int st = 256; st > 0; st >>= 1) {
        if (t < st) red[t] += red[t + st];
        __syncthreads();
    }
    const float var = red[0] * (1.0f / HDI);
    x[(long)m * HDI + t] = d * rsqrtf(var + 1e-5f) * gin[t] + bein[t];
}

// ---------------------------------------------------------------------------
// Weight transpose + fp32 -> bf16:  Wt[n,k] = W[k,n]
// ---------------------------------------------------------------------------
__global__ void transpose_k(const float* __restrict__ W, bf16* __restrict__ Wt, int K, int N) {
    const long t = (long)blockIdx.x * 256 + threadIdx.x;
    if (t >= (long)K * N) return;
    const int k = (int)(t / N);
    const int n = (int)(t % N);
    Wt[(long)n * K + k] = (bf16)W[t];
}

__global__ void to_bf16_k(const float* __restrict__ x, bf16* __restrict__ y, int n) {
    const int i = blockIdx.x * 256 + threadIdx.x;
    if (i < n) y[i] = (bf16)x[i];
}

__global__ void zero_k(float* __restrict__ p, int n) {
    const int i = blockIdx.x * 256 + threadIdx.x;
    if (i < n) p[i] = 0.0f;
}

// Final head matvec (256 -> 1) with atom_name selection
__global__ void head_final(const bf16* __restrict__ t2, const float* __restrict__ hW3,
                           const float* __restrict__ hb3, const int* __restrict__ aname,
                           float* __restrict__ preds, int head) {
    const int m = blockIdx.x * 256 + threadIdx.x;
    if (m >= MI) return;
    if (aname[m] != head) return;
    float acc = hb3[0];
    for (int k = 0; k < HHI; ++k) acc += (float)t2[(long)m * HHI + k] * hW3[k];
    preds[m] = acc;
}

// ---------------------------------------------------------------------------
// Workspace layout (bytes)
// ---------------------------------------------------------------------------
constexpr size_t OFF_XF  = 0;                                        // f32 [2048,512]
constexpr size_t OFF_HB  = OFF_XF  + (size_t)MI * HDI * 4;           // bf16 [2048,512]
constexpr size_t OFF_QB  = OFF_HB  + (size_t)MI * HDI * 2;
constexpr size_t OFF_KB  = OFF_QB  + (size_t)MI * HDI * 2;
constexpr size_t OFF_VB  = OFF_KB  + (size_t)MI * HDI * 2;
constexpr size_t OFF_OB  = OFF_VB  + (size_t)MI * HDI * 2;
constexpr size_t OFF_SC  = OFF_OB  + (size_t)MI * HDI * 2;           // f32 [32,512,512]
constexpr size_t OFF_AT  = OFF_SC  + (size_t)BI * NHI * NI * NI * 4; // bf16 [32,512,512]
constexpr size_t OFF_TB  = OFF_AT  + (size_t)BI * NHI * NI * NI * 2; // bf16 [2048,2048]
constexpr size_t OFF_VT  = OFF_TB  + (size_t)MI * FFI * 2;           // bf16 [32,64,512]
constexpr size_t OFF_W0  = OFF_VT  + (size_t)BI * NHI * DHI * NI * 2;
constexpr size_t OFF_W1  = OFF_W0  + (size_t)HDI * HDI * 2;
constexpr size_t OFF_W2  = OFF_W1  + (size_t)HDI * HDI * 2;
constexpr size_t OFF_W3  = OFF_W2  + (size_t)HDI * HDI * 2;
constexpr size_t OFF_WF1 = OFF_W3  + (size_t)HDI * HDI * 2;
constexpr size_t OFF_WF2 = OFF_WF1 + (size_t)HDI * FFI * 2;

extern "C" void kernel_launch(void* const* d_in, const int* in_sizes, int n_in,
                              void* d_out, int out_size, void* d_ws, size_t ws_size,
                              hipStream_t stream) {
    const int*   atom_type = (const int*)d_in[0];
    const int*   atom_name = (const int*)d_in[1];
    const int*   res_type  = (const int*)d_in[2];
    const int*   ss_type   = (const int*)d_in[3];
    const int*   relpos    = (const int*)d_in[4];
    const float* coords    = (const float*)d_in[5];
    const float* phi       = (const float*)d_in[6];
    const float* psi       = (const float*)d_in[7];
    const float* cs_input  = (const float*)d_in[8];
    const float* dist      = (const float*)d_in[9];
    const unsigned char* mask = (const unsigned char*)d_in[10];
    const float* Eat  = (const float*)d_in[11];
    const float* Ean  = (const float*)d_in[12];
    const float* Er   = (const float*)d_in[13];
    const float* Ess  = (const float*)d_in[14];
    const float* Ep   = (const float*)d_in[15];
    const float* Wc   = (const float*)d_in[16];
    const float* bc   = (const float*)d_in[17];
    const float* Win  = (const float*)d_in[18];
    const float* bin  = (const float*)d_in[19];
    const float* gin  = (const float*)d_in[20];
    const float* bein = (const float*)d_in[21];
    const float* centers = (const float*)d_in[22];
    const float* widths  = (const float*)d_in[23];
    const float* Wq  = (const float*)d_in[24];
    const float* Wk  = (const float*)d_in[25];
    const float* Wv  = (const float*)d_in[26];
    const float* Wo  = (const float*)d_in[27];
    const float* Wd  = (const float*)d_in[28];
    const float* Wf1 = (const float*)d_in[29];
    const float* Wf2 = (const float*)d_in[30];
    const float* bq  = (const float*)d_in[31];
    const float* bk  = (const float*)d_in[32];
    const float* bv  = (const float*)d_in[33];
    const float* bo  = (const float*)d_in[34];
    const float* bd  = (const float*)d_in[35];
    const float* bf1 = (const float*)d_in[36];
    const float* bf2 = (const float*)d_in[37];
    const float* g1  = (const float*)d_in[38];
    const float* b1  = (const float*)d_in[39];
    const float* g2  = (const float*)d_in[40];
    const float* b2  = (const float*)d_in[41];
    const float* hW1 = (const float*)d_in[42];
    const float* hb1 = (const float*)d_in[43];
    const float* hW2 = (const float*)d_in[44];
    const float* hb2 = (const float*)d_in[45];
    const float* hW3 = (const float*)d_in[46];
    const float* hb3 = (const float*)d_in[47];

    char* ws = (char*)d_ws;
    float* xF  = (float*)(ws + OFF_XF);
    bf16*  hB  = (bf16*)(ws + OFF_HB);
    bf16*  qB  = (bf16*)(ws + OFF_QB);
    bf16*  kB  = (bf16*)(ws + OFF_KB);
    bf16*  vB  = (bf16*)(ws + OFF_VB);
    bf16*  oB  = (bf16*)(ws + OFF_OB);
    float* scF = (float*)(ws + OFF_SC);
    bf16*  atB = (bf16*)(ws + OFF_AT);
    bf16*  tB  = (bf16*)(ws + OFF_TB);
    bf16*  vT  = (bf16*)(ws + OFF_VT);
    bf16*  W0  = (bf16*)(ws + OFF_W0);
    bf16*  W1  = (bf16*)(ws + OFF_W1);
    bf16*  W2  = (bf16*)(ws + OFF_W2);
    bf16*  W3  = (bf16*)(ws + OFF_W3);
    bf16*  WF1 = (bf16*)(ws + OFF_WF1);
    bf16*  WF2 = (bf16*)(ws + OFF_WF2);

    float* preds = (float*)d_out;

    // 1) embedding + input projection + LN
    embed_k<<<MI, 512, 0, stream>>>(atom_type, atom_name, res_type, ss_type, relpos,
                                    coords, phi, psi, cs_input,
                                    Eat, Ean, Er, Ess, Ep, Wc, bc, Win, bin, gin, bein, xF);

    const float scale = 0.125f; // 1/sqrt(DH)

    for (int l = 0; l < LI; ++l) {
        const float* Wq_l  = Wq  + (size_t)l * HDI * HDI;
        const float* Wk_l  = Wk  + (size_t)l * HDI * HDI;
        const float* Wv_l  = Wv  + (size_t)l * HDI * HDI;
        const float* Wo_l  = Wo  + (size_t)l * HDI * HDI;
        const float* Wd_l  = Wd  + (size_t)l * NKI * NHI;
        const float* Wf1_l = Wf1 + (size_t)l * HDI * FFI;
        const float* Wf2_l = Wf2 + (size_t)l * FFI * HDI;

        // weight transposes -> bf16 [N,K]
        transpose_k<<<(HDI * HDI) / 256, 256, 0, stream>>>(Wq_l, W0, HDI, HDI);
        transpose_k<<<(HDI * HDI) / 256, 256, 0, stream>>>(Wk_l, W1, HDI, HDI);
        transpose_k<<<(HDI * HDI) / 256, 256, 0, stream>>>(Wv_l, W2, HDI, HDI);
        transpose_k<<<(HDI * HDI) / 256, 256, 0, stream>>>(Wo_l, W3, HDI, HDI);
        transpose_k<<<(HDI * FFI) / 256, 256, 0, stream>>>(Wf1_l, WF1, HDI, FFI);
        transpose_k<<<(FFI * HDI) / 256, 256, 0, stream>>>(Wf2_l, WF2, FFI, HDI);

        // LN1 -> h (bf16)
        ln_k<<<MI, 512, 0, stream>>>(xF, g1 + l * HDI, b1 + l * HDI, hB);

        // q, k, v projections (WMMA, 32x64 tiles): 512 tiles -> 64 blocks
        gemm_bf16out<<<64, 256, 0, stream>>>(hB, W0, bq + l * HDI, qB, MI, HDI, HDI, 0);
        gemm_bf16out<<<64, 256, 0, stream>>>(hB, W1, bk + l * HDI, kB, MI, HDI, HDI, 0);
        gemm_bf16out<<<64, 256, 0, stream>>>(hB, W2, bv + l * HDI, vB, MI, HDI, HDI, 0);

        // scores = scale * q k^T  (WMMA, batched over b,h): 128 tiles/z -> 16 blocks
        attn_scores<<<dim3(16, BI * NHI), 256, 0, stream>>>(qB, kB, scF, scale);

        // += RBF distance bias
        bias_add<<<(BI * NI * NI) / 256, 256, 0, stream>>>(dist, centers, widths, Wd_l,
                                                           bd + l * NHI, scF);

        // masked softmax -> bf16 attn
        softmax_k<<<BI * NHI * NI, 512, 0, stream>>>(scF, mask, atB);

        // transpose V for contiguous attn@V loads
        vt_k<<<(BI * NHI * DHI * NI) / 256, 256, 0, stream>>>(vB, vT);

        // o = attn @ v (WMMA, batched): 16 tiles/z -> 2 blocks
        attn_out<<<dim3(2, BI * NHI), 256, 0, stream>>>(atB, vT, oB);

        // x += (o @ Wo + bo) * mask
        gemm_f32resid<<<64, 256, 0, stream>>>(oB, W3, bo + l * HDI, xF, mask, MI, HDI, HDI);

        // LN2 -> h2; FFN
        ln_k<<<MI, 512, 0, stream>>>(xF, g2 + l * HDI, b2 + l * HDI, hB);
        gemm_bf16out<<<256, 256, 0, stream>>>(hB, WF1, bf1 + l * FFI, tB, MI, FFI, HDI, 1);
        gemm_f32resid<<<64, 256, 0, stream>>>(tB, WF2, bf2 + l * HDI, xF, nullptr, MI, HDI, FFI);
    }

    // heads
    to_bf16_k<<<(MI * HDI) / 256, 256, 0, stream>>>(xF, hB, MI * HDI);
    zero_k<<<MI / 256, 256, 0, stream>>>(preds, MI);
    for (int i = 0; i < NBI; ++i) {
        transpose_k<<<(HDI * HDI) / 256, 256, 0, stream>>>(hW1 + (size_t)i * HDI * HDI, W0, HDI, HDI);
        transpose_k<<<(HDI * HHI) / 256, 256, 0, stream>>>(hW2 + (size_t)i * HDI * HHI, W1, HDI, HHI);
        gemm_bf16out<<<64, 256, 0, stream>>>(hB, W0, hb1 + i * HDI, qB, MI, HDI, HDI, 1);
        gemm_bf16out<<<32, 256, 0, stream>>>(qB, W1, hb2 + i * HHI, kB, MI, HHI, HDI, 1);
        head_final<<<MI / 256, 256, 0, stream>>>(kB, hW3 + (size_t)i * HHI, hb3 + i,
                                                 atom_name, preds, i);
    }
}
